// BootlegM2E_48387101557004
// MI455X (gfx1250) — matrix operand order repaired and checked by reading, not compiled, not run on hardware
//
#include <hip/hip_runtime.h>
#include <hip/hip_bf16.h>
#include <stdint.h>

// ---------------------------------------------------------------------------
// Problem constants (from the reference)
// ---------------------------------------------------------------------------
#define BV     16      // batch
#define NNW    256     // sentence length N
#define MMW    16      // mentions M
#define KKW    32      // candidates K
#define HH     768     // hidden
#define NHEADS 12
#define DH     64      // head dim
#define FFD    3072
#define MKD    512     // M*K
#define NEGV   (-1e9f)

typedef __bf16 bf16;
typedef __attribute__((ext_vector_type(16))) __bf16 v16bf;
typedef __attribute__((ext_vector_type(8)))  __bf16 v8bf;
typedef __attribute__((ext_vector_type(8)))  float  v8f;

// ---------------------------------------------------------------------------
// WMMA fragment loaders (CDNA5 gfx1250, wave32).  Operands are bf16 in memory.
// A fragment: 16x32 (MxK). lanes 0-15 row M=lane: elems[0..15]=K 0..15;
// lanes 16-31 same rows with K 16..31.
// ---------------------------------------------------------------------------
__device__ __forceinline__ v16bf load_a_frag(const bf16* A, int lda,
                                             int m0, int kb, int lane) {
  int r    = m0 + (lane & 15);
  int koff = (lane >> 4) << 4;
  const bf16* p = A + (long long)r * lda + kb + koff;
  v8bf lo = *(const v8bf*)p;
  v8bf hi = *(const v8bf*)(p + 8);
  return __builtin_shufflevector(lo, hi, 0, 1, 2, 3, 4, 5, 6, 7,
                                 8, 9, 10, 11, 12, 13, 14, 15);
}

// B fragment: 32x16 (KxN) from W[N,K] row-major (TN GEMM). Lane n<16 holds
// column n with K 0..15 contiguous; lane>=16 same column, K 16..31.
__device__ __forceinline__ v16bf load_b_frag(const bf16* B, int ldb,
                                             int n0, int kb, int lane) {
  int n    = n0 + (lane & 15);
  int koff = (lane >> 4) << 4;
  const bf16* p = B + (long long)n * ldb + kb + koff;
  v8bf lo = *(const v8bf*)p;
  v8bf hi = *(const v8bf*)(p + 8);
  return __builtin_shufflevector(lo, hi, 0, 1, 2, 3, 4, 5, 6, 7,
                                 8, 9, 10, 11, 12, 13, 14, 15);
}

#define WMMA_BF16(a, b, c) \
  __builtin_amdgcn_wmma_f32_16x16x32_bf16(false, (a), false, (b), (short)0, (c), false, false)

// ---------------------------------------------------------------------------
// Generic batched TN GEMM: C = relu?( alpha*(A x B^T) + bias + beta*R )
// A: (Mr x Kd) bf16, B: (Nc x Kd) bf16, batched over gridDim.z = nb*heads.
// C written f32 or bf16, optionally transposed (C[col*ldc+row]).
// REQUIREMENT (holds for every GEMM in this model): Mr % 32 == 0,
// Nc % 64 == 0, Kd % 32 == 0 — so the hot loop carries no bounds checks.
// ---------------------------------------------------------------------------
struct GP {
  const bf16* A;  long long Ab, Ah;  int lda;
  const bf16* B;  long long Bb, Bh;  int ldb;
  void*       C;  long long Cb, Ch;  int ldc;
  const float* bias;
  const float* R; long long Rb, Rh;  int ldr;
  int   Mr, Nc, Kd, heads;
  float alpha, beta;
  int   relu, cbf16, ctrans;
};

__device__ __forceinline__ void store_tile(const GP& p, char* C, const float* R,
                                           v8f acc, int r0, int col) {
#pragma unroll
  for (int r = 0; r < 8; ++r) {
    int row = r0 + r;
    float v = p.alpha * acc[r];
    if (p.bias) v += p.bias[col];
    if (R)      v += p.beta * R[(long long)row * p.ldr + col];
    if (p.relu) v = fmaxf(v, 0.f);
    long long idx = p.ctrans ? ((long long)col * p.ldc + row)
                             : ((long long)row * p.ldc + col);
    if (p.cbf16) ((bf16*)C)[idx] = (bf16)v;
    else         ((float*)C)[idx] = v;
  }
}

__global__ __launch_bounds__(256) void wmma_gemm_k(GP p) {
  const int z = blockIdx.z;
  const int b = z / p.heads, h = z % p.heads;
  const bf16* A = p.A + (long long)b * p.Ab + (long long)h * p.Ah;
  const bf16* B = p.B + (long long)b * p.Bb + (long long)h * p.Bh;
  char* C = (char*)p.C + ((long long)b * p.Cb + (long long)h * p.Ch)
                           * (p.cbf16 ? 2 : 4);
  const float* R = p.R ? p.R + (long long)b * p.Rb + (long long)h * p.Rh : nullptr;

  const int tn_n = p.Nc >> 6;              // 64-wide N tiles per wave
  const int tn_m = p.Mr >> 5;              // 32-tall M tiles per wave
  int w = blockIdx.x * (blockDim.x >> 5) + (threadIdx.x >> 5);
  if (w >= tn_m * tn_n) return;            // only possible in the last block
  const int tm = w / tn_n, tn = w % tn_n;
  const int m0 = tm << 5, n0 = tn << 6;
  const int lane = threadIdx.x & 31;

  v8f acc[2][4] = {};
  for (int kb = 0; kb < p.Kd; kb += 32) {
    // speculative prefetch of the next K-slab (global_prefetch_b8);
    // harmless past the end — dropped on translation failure.
    __builtin_prefetch(A + (long long)(m0 + (lane & 15)) * p.lda + kb + 32, 0, 1);
    v16bf a0 = load_a_frag(A, p.lda, m0,      kb, lane);
    v16bf a1 = load_a_frag(A, p.lda, m0 + 16, kb, lane);
#pragma unroll
    for (int j = 0; j < 4; ++j) {
      v16bf bf = load_b_frag(B, p.ldb, n0 + 16 * j, kb, lane);
      acc[0][j] = WMMA_BF16(a0, bf, acc[0][j]);
      acc[1][j] = WMMA_BF16(a1, bf, acc[1][j]);
    }
  }
  const int col0 = n0 + (lane & 15);
  const int r0   = (lane >> 4) << 3;
#pragma unroll
  for (int j = 0; j < 4; ++j) {
    store_tile(p, C, R, acc[0][j], m0 + r0,      col0 + 16 * j);
    store_tile(p, C, R, acc[1][j], m0 + 16 + r0, col0 + 16 * j);
  }
}

// ---------------------------------------------------------------------------
// Block reductions (wave32)
// ---------------------------------------------------------------------------
__device__ __forceinline__ float blk_sum(float v) {
  __shared__ float sh[8];
  for (int o = 16; o > 0; o >>= 1) v += __shfl_xor(v, o, 32);
  int w = threadIdx.x >> 5;
  if ((threadIdx.x & 31) == 0) sh[w] = v;
  __syncthreads();
  int nw = (blockDim.x + 31) >> 5;
  float r = 0.f;
  for (int i = 0; i < nw; ++i) r += sh[i];
  __syncthreads();
  return r;
}
__device__ __forceinline__ float blk_max(float v) {
  __shared__ float sh[8];
  for (int o = 16; o > 0; o >>= 1) v = fmaxf(v, __shfl_xor(v, o, 32));
  int w = threadIdx.x >> 5;
  if ((threadIdx.x & 31) == 0) sh[w] = v;
  __syncthreads();
  int nw = (blockDim.x + 31) >> 5;
  float r = -3.4e38f;
  for (int i = 0; i < nw; ++i) r = fmaxf(r, sh[i]);
  __syncthreads();
  return r;
}

// ---------------------------------------------------------------------------
// LayerNorm: one block per row; bf16 output (all LN outputs feed GEMM A-ops)
// ---------------------------------------------------------------------------
__global__ __launch_bounds__(256) void layernorm_k(const float* X, const float* g,
                                                   const float* bb, bf16* Y, int H) {
  long long row = blockIdx.x;
  const float* x = X + row * (long long)H;
  bf16*        y = Y + row * (long long)H;
  float s = 0.f, sq = 0.f;
  for (int h = threadIdx.x; h < H; h += blockDim.x) { float v = x[h]; s += v; sq += v * v; }
  float mu  = blk_sum(s) / H;
  float var = blk_sum(sq) / H - mu * mu;
  float rs  = rsqrtf(var + 1e-5f);
  for (int h = threadIdx.x; h < H; h += blockDim.x)
    y[h] = (bf16)((x[h] - mu) * rs * g[h] + bb[h]);
}

// NormAndSum: out = ln(x0)+ln(x1)+ln(x2) (shared g/b); x2 row = row/div2. f32 out.
__global__ __launch_bounds__(256) void norm3_sum_k(const float* x0, const float* x1,
                                                   const float* x2, int div2,
                                                   const float* g, const float* bb,
                                                   float* out, int H) {
  long long row = blockIdx.x;
  const float* p0 = x0 + row * (long long)H;
  const float* p1 = x1 + row * (long long)H;
  const float* p2 = x2 + (row / div2) * (long long)H;
  float s0 = 0, q0 = 0, s1 = 0, q1 = 0, s2 = 0, q2 = 0;
  for (int h = threadIdx.x; h < H; h += blockDim.x) {
    float a = p0[h], b = p1[h], c = p2[h];
    s0 += a; q0 += a * a; s1 += b; q1 += b * b; s2 += c; q2 += c * c;
  }
  s0 = blk_sum(s0); q0 = blk_sum(q0);
  s1 = blk_sum(s1); q1 = blk_sum(q1);
  s2 = blk_sum(s2); q2 = blk_sum(q2);
  float inv = 1.f / H;
  float mu0 = s0 * inv, mu1 = s1 * inv, mu2 = s2 * inv;
  float r0 = rsqrtf(q0 * inv - mu0 * mu0 + 1e-5f);
  float r1 = rsqrtf(q1 * inv - mu1 * mu1 + 1e-5f);
  float r2 = rsqrtf(q2 * inv - mu2 * mu2 + 1e-5f);
  float* o = out + row * (long long)H;
  for (int h = threadIdx.x; h < H; h += blockDim.x)
    o[h] = g[h] * ((p0[h] - mu0) * r0 + (p1[h] - mu1) * r1 + (p2[h] - mu2) * r2) + 3.f * bb[h];
}

// ---------------------------------------------------------------------------
// Masked softmax: f32 scores in, bf16 probs out.
// e2e!=0 adds -1e9 where (q/32)==(k/32); pad overrides to exactly -1e9.
// ---------------------------------------------------------------------------
__global__ __launch_bounds__(256) void softmax_mask_k(float* S, bf16* Pr, const int* pad,
                                                      int Lq, int Lk, int heads, int e2e) {
  long long row = blockIdx.x;
  int q = (int)(row % Lq);
  int b = (int)((row / Lq) / heads);
  float* s = S + row * (long long)Lk;
  bf16*  pr = Pr + row * (long long)Lk;
  int qb = q >> 5;
  float lmax = -3.4e38f;
  for (int k = threadIdx.x; k < Lk; k += blockDim.x) {
    float v = s[k];
    if (e2e && ((k >> 5) == qb)) v += NEGV;
    if (pad && pad[(long long)b * Lk + k]) v = NEGV;
    s[k] = v;
    lmax = fmaxf(lmax, v);
  }
  float mx = blk_max(lmax);
  float lsum = 0.f;
  for (int k = threadIdx.x; k < Lk; k += blockDim.x) {
    float e = __expf(s[k] - mx);
    s[k] = e;
    lsum += e;
  }
  float inv = 1.f / blk_sum(lsum);
  for (int k = threadIdx.x; k < Lk; k += blockDim.x)
    pr[k] = (bf16)(s[k] * inv);
}

// ---------------------------------------------------------------------------
// Mention attention (Lq=1, Lk=32): one wave per (bm, head); shuffle softmax.
// ---------------------------------------------------------------------------
__global__ __launch_bounds__(32) void me_attn_k(const bf16* Q, const bf16* Km,
                                                const bf16* V, const int* padAdj,
                                                bf16* O) {
  int z = blockIdx.x;                 // 256*12
  int bm = z / NHEADS, h = z % NHEADS;
  int lane = threadIdx.x;             // one key per lane
  const bf16* q  = Q  + (long long)bm * HH + h * DH;
  const bf16* kr = Km + ((long long)bm * KKW + lane) * HH + h * DH;
  float s = 0.f;
#pragma unroll 8
  for (int d = 0; d < DH; ++d) s += (float)q[d] * (float)kr[d];
  s *= 0.125f;                        // 1/sqrt(64)
  if (padAdj[bm * KKW + lane]) s = NEGV;
  float mx = s;
  for (int o = 16; o > 0; o >>= 1) mx = fmaxf(mx, __shfl_xor(mx, o, 32));
  float e = __expf(s - mx);
  float sum = e;
  for (int o = 16; o > 0; o >>= 1) sum += __shfl_xor(sum, o, 32);
  float pl = e / sum;
  float o0 = 0.f, o1 = 0.f;
  for (int k = 0; k < KKW; ++k) {
    float pk = __shfl(pl, k, 32);
    const bf16* vr = V + ((long long)bm * KKW + k) * HH + h * DH;
    o0 += pk * (float)vr[lane];
    o1 += pk * (float)vr[lane + 32];
  }
  bf16* op = O + (long long)bm * HH + h * DH;
  op[lane]      = (bf16)o0;
  op[lane + 32] = (bf16)o1;
}

// ---------------------------------------------------------------------------
// Conversion / utility kernels
// ---------------------------------------------------------------------------
__global__ void cvt_bf16_k(const float* X, bf16* Y, long long n) {
  long long i = (long long)blockIdx.x * blockDim.x + threadIdx.x;
  if (i < n) Y[i] = (bf16)X[i];
}

// Batched transpose+convert: X (nb, rows, cols) f32 -> Y (nb, cols, rows) bf16
__global__ void cvtT_bf16_k(const float* X, bf16* Y, int rows, int cols, long long total) {
  long long i = (long long)blockIdx.x * blockDim.x + threadIdx.x;
  if (i >= total) return;
  long long per = (long long)rows * cols;
  long long b = i / per, rem = i % per;
  int r = (int)(rem / cols), c = (int)(rem % cols);
  Y[b * per + (long long)c * rows + r] = (bf16)X[i];
}

__global__ void mention_gather_k(const float* sent, const int* sidx, const int* eidx,
                                 float* mention) {
  long long i = (long long)blockIdx.x * blockDim.x + threadIdx.x;
  if (i >= (long long)BV * MMW * HH) return;
  int bm = (int)(i / HH), h = (int)(i % HH);
  int b = bm / MMW;
  int i0 = sidx[bm], i1 = eidx[bm];
  int s0 = i0 < 0 ? 0 : i0, s1 = i1 < 0 ? 0 : i1;
  float v0 = (i0 < 0) ? 0.f : sent[((long long)b * NNW + s0) * HH + h];
  float v1 = (i1 < 0) ? 0.f : sent[((long long)b * NNW + s1) * HH + h];
  mention[i] = 0.5f * (v0 + v1);
}

__global__ void aux_masks_k(const int* entMask, int* allPad, int* nonNull, int* padAdj) {
  int bm = blockIdx.x * blockDim.x + threadIdx.x;
  if (bm >= BV * MMW) return;
  int cnt = 0;
  for (int k = 0; k < KKW; ++k) cnt += (entMask[bm * KKW + k] != 0);
  int ap = (cnt == KKW);
  allPad[bm]  = ap;
  nonNull[bm] = ((KKW - cnt) != 0);
  for (int k = 0; k < KKW; ++k)
    padAdj[bm * KKW + k] = ap ? 0 : (entMask[bm * KKW + k] != 0);
}

__global__ void aux_single_k(const int* nonNull, int* single) {
  int b = threadIdx.x;
  if (b >= BV) return;
  int s = 0;
  for (int m = 0; m < MMW; ++m) s += nonNull[b * MMW + m];
  single[b] = (s == 1);
}

__global__ void zero_rows_k(float* X, const int* flag, int rowsPerFlag, int H,
                            long long rows) {
  long long i = (long long)blockIdx.x * blockDim.x + threadIdx.x;
  if (i >= rows * H) return;
  long long row = i / H;
  if (flag[row / rowsPerFlag]) X[i] = 0.f;
}

__global__ __launch_bounds__(256) void rowdot_k(const float* T, const float* w,
                                                const float* b1, float* out,
                                                int rows, int H) {
  int row = blockIdx.x * 8 + (threadIdx.x >> 5);
  if (row >= rows) return;
  int lane = threadIdx.x & 31;
  float s = 0.f;
  for (int h = lane; h < H; h += 32) s += T[(long long)row * H + h] * w[h];
  for (int o = 16; o > 0; o >>= 1) s += __shfl_xor(s, o, 32);
  if (lane == 0) out[row] = s + b1[0];
}

__global__ void max2_k(const float* a, const float* b, float* o, long long n) {
  long long i = (long long)blockIdx.x * blockDim.x + threadIdx.x;
  if (i < n) o[i] = fmaxf(a[i], b[i]);
}

// ---------------------------------------------------------------------------
// Host orchestration
// ---------------------------------------------------------------------------
struct BPidx { int ln_q_g, ln_q_b, ln_x_g, ln_x_b, Wi, bi, Wo, bo, ln_f_g, ln_f_b, W1, b1, W2, b2; };
struct StageIdx { BPidx ew, se, me; int comb_g, comb_b, p_W0, p_b0, p_W1, p_b1; };

extern "C" void kernel_launch(void* const* d_in, const int* in_sizes, int n_in,
                              void* d_out, int out_size, void* d_ws, size_t ws_size,
                              hipStream_t stream) {
  (void)in_sizes; (void)n_in; (void)out_size; (void)ws_size;

  // ---- input index table (params dict in insertion order, then tensors) ----
  StageIdx sp[3];
  int idx = 0;
  for (int s = 0; s < 3; ++s) {
    auto fill = [&](BPidx& bp, bool has_x) {
      bp.ln_q_g = idx++; bp.ln_q_b = idx++;
      if (has_x) { bp.ln_x_g = idx++; bp.ln_x_b = idx++; } else { bp.ln_x_g = bp.ln_x_b = -1; }
      bp.Wi = idx++; bp.bi = idx++; bp.Wo = idx++; bp.bo = idx++;
      bp.ln_f_g = idx++; bp.ln_f_b = idx++;
      bp.W1 = idx++; bp.b1 = idx++; bp.W2 = idx++; bp.b2 = idx++;
    };
    fill(sp[s].ew, true); fill(sp[s].se, false); fill(sp[s].me, true);
    sp[s].comb_g = idx++; sp[s].comb_b = idx++;
    if (s < 2) { sp[s].p_W0 = idx++; sp[s].p_b0 = idx++; sp[s].p_W1 = idx++; sp[s].p_b1 = idx++; }
    else { sp[s].p_W0 = sp[s].p_b0 = sp[s].p_W1 = sp[s].p_b1 = -1; }
  }
  const int iSent = idx++, iSentMask = idx++, iEnt = idx++, iEntMask = idx++;
  const int iStart = idx++, iEnd = idx++, iKg = idx++;

  auto P = [&](int i) -> const float* { return (const float*)d_in[i]; };
  auto I = [&](int i) -> const int*   { return (const int*)d_in[i]; };

  // ---- workspace bump allocator ----
  size_t off = 0;
  auto wsf = [&](size_t elems) -> float* {
    float* p = (float*)((char*)d_ws + off);
    off += (elems * sizeof(float) + 255) & ~(size_t)255;
    return p;
  };
  auto wsh = [&](size_t elems) -> bf16* {
    bf16* p = (bf16*)((char*)d_ws + off);
    off += (elems * sizeof(bf16) + 255) & ~(size_t)255;
    return p;
  };
  auto wsi = [&](size_t elems) -> int* {
    int* p = (int*)((char*)d_ws + off);
    off += (elems * sizeof(int) + 255) & ~(size_t)255;
    return p;
  };

  const size_t QN  = (size_t)BV * MKD * HH;           // 6,291,456
  const size_t MEN = (size_t)BV * MMW * HH;           // 196,608
  const long long SC = (long long)BV * NHEADS * MKD * MKD;

  // f32 buffers
  float* qA     = wsf(QN);
  float* qB     = wsf(QN);
  float* mention= wsf(MEN);
  float* scores = wsf((size_t)SC);
  float* ybuf   = wsf(QN);
  float* ew_out = wsf(QN);
  float* se_out = wsf(QN);
  float* me_y   = wsf(MEN);
  float* me_out = wsf(MEN);
  float* ctx    = wsf(QN);
  float* t0     = wsf(QN);
  float* sc0    = wsf(BV * MKD);
  float* sc1    = wsf(BV * MKD);
  // bf16 buffers
  bf16* qn     = wsh(QN);
  bf16* xn     = wsh(QN);
  bf16* Qp     = wsh(QN);
  bf16* Kp     = wsh(QN);
  bf16* Vp     = wsh(QN);            // me block (row-major V)
  bf16* Vt     = wsh(QN);            // ew/se (per-batch transposed V)
  bf16* probs  = wsh((size_t)SC);
  bf16* attnO  = wsh(QN);
  bf16* hn     = wsh(QN);
  bf16* ff1    = wsh((size_t)BV * MKD * FFD);
  bf16* me_qn  = wsh(MEN);
  bf16* me_Q   = wsh(MEN);
  bf16* me_O   = wsh(MEN);
  bf16* me_hn  = wsh(MEN);
  bf16* me_ff1 = wsh((size_t)BV * MMW * FFD);
  bf16* bfW    = wsh((size_t)3 * HH * HH);  // sized for Wi; W1/W2 (FFD*HH) are larger
  bf16* kg_bf  = wsh((size_t)BV * MKD * MKD);
  bf16* ctxT   = wsh(QN);
  bf16* ctx_bf = wsh(QN);
  bf16* ckg_bf = wsh(QN);
  // aux ints
  int* allPad  = wsi(BV * MMW);
  int* nonNull = wsi(BV * MMW);
  int* single  = wsi(BV);
  int* padAdj  = wsi(BV * MMW * KKW);

  // ---- launch helpers ----
  auto gemm = [&](const bf16* A, long long Ab, long long Ah, int lda,
                  const bf16* B, long long Bb, long long Bh, int ldb,
                  void* C, long long Cb, long long Ch, int ldc,
                  const float* bias,
                  const float* R, long long Rb, long long Rh, int ldr,
                  int Mr, int Nc, int Kd, int nb, int heads,
                  float alpha, float beta, int relu, int cbf16, int ctrans) {
    GP p{A, Ab, Ah, lda, B, Bb, Bh, ldb, C, Cb, Ch, ldc, bias, R, Rb, Rh, ldr,
         Mr, Nc, Kd, heads, alpha, beta, relu, cbf16, ctrans};
    int tiles = (Mr / 32) * (Nc / 64);
    dim3 grid((tiles + 7) / 8, 1, nb * heads);
    wmma_gemm_k<<<grid, 256, 0, stream>>>(p);
  };
  auto LN = [&](const float* X, int gi, int bi_, bf16* Y, long long rows) {
    layernorm_k<<<dim3((unsigned)rows), 256, 0, stream>>>(X, P(gi), P(bi_), Y, HH);
  };
  auto CVT = [&](const float* X, bf16* Y, long long n) {
    cvt_bf16_k<<<dim3((unsigned)((n + 255) / 256)), 256, 0, stream>>>(X, Y, n);
  };

  // ---- precompute: mention, masks, kg->bf16 (constant across stages) ----
  {
    long long n = (long long)BV * MMW * HH;
    mention_gather_k<<<dim3((unsigned)((n + 255) / 256)), 256, 0, stream>>>(
        P(iSent), I(iStart), I(iEnd), mention);
    aux_masks_k<<<1, 256, 0, stream>>>(I(iEntMask), allPad, nonNull, padAdj);
    aux_single_k<<<1, 32, 0, stream>>>(nonNull, single);
    CVT(P(iKg), kg_bf, (long long)BV * MKD * MKD);
  }

  const float* cur_q = P(iEnt);              // query = entity_embedding (B, MK, H)
  const int ROWS_Q = BV * MKD;               // 8192

  // standard attn_block (ew / se): query is (B,512,H)
  auto run_std_block = [&](const BPidx& bp, const float* q_in, const float* kv_in,
                           bool self_attn, int Lk, const int* pad, int e2e, float* out) {
    const int rows_k = BV * Lk;
    LN(q_in, bp.ln_q_g, bp.ln_q_b, qn, ROWS_Q);
    const bf16* xnp = qn;
    if (!self_attn) { LN(kv_in, bp.ln_x_g, bp.ln_x_b, xn, rows_k); xnp = xn; }
    // weights -> bf16 (Wi covers Q,K,V slices)
    CVT(P(bp.Wi), bfW, (long long)3 * HH * HH);
    const float* bi = P(bp.bi);
    // Q/K projections (row-major bf16 out)
    gemm(qn, 0, 0, HH, bfW, 0, 0, HH, Qp, 0, 0, HH, bi,
         nullptr, 0, 0, 0, ROWS_Q, HH, HH, 1, 1, 1.f, 0.f, 0, 1, 0);
    gemm(xnp, 0, 0, HH, bfW + (size_t)HH * HH, 0, 0, HH, Kp, 0, 0, HH, bi + HH,
         nullptr, 0, 0, 0, rows_k, HH, HH, 1, 1, 1.f, 0.f, 0, 1, 0);
    // V projection stored transposed per batch: Vt[b][hcol][k], ldc=Lk
    gemm(xnp, (long long)Lk * HH, 0, HH,
         bfW + (size_t)2 * HH * HH, 0, 0, HH,
         Vt, (long long)HH * Lk, 0, Lk, bi + 2 * HH,
         nullptr, 0, 0, 0, Lk, HH, HH, BV, 1, 1.f, 0.f, 0, 1, 1);
    // scores = (Q K^T)/8, batched over (b,h), f32
    gemm(Qp, (long long)MKD * HH, DH, HH,
         Kp, (long long)Lk * HH,  DH, HH,
         scores, (long long)NHEADS * MKD * Lk, (long long)MKD * Lk, Lk,
         nullptr, nullptr, 0, 0, 0,
         MKD, Lk, DH, BV, NHEADS, 0.125f, 0.f, 0, 0, 0);
    softmax_mask_k<<<dim3((unsigned)(BV * NHEADS * MKD)), 256, 0, stream>>>(
        scores, probs, pad, MKD, Lk, NHEADS, e2e);
    // attnO = P @ V  ==  TN gemm against transposed V
    gemm(probs, (long long)NHEADS * MKD * Lk, (long long)MKD * Lk, Lk,
         Vt, (long long)HH * Lk, (long long)DH * Lk, Lk,
         attnO, (long long)MKD * HH, DH, HH,
         nullptr, nullptr, 0, 0, 0,
         MKD, DH, Lk, BV, NHEADS, 1.f, 0.f, 0, 1, 0);
    // y = q + attnO @ Wo^T + bo
    CVT(P(bp.Wo), bfW, (long long)HH * HH);
    gemm(attnO, 0, 0, HH, bfW, 0, 0, HH, ybuf, 0, 0, HH, P(bp.bo),
         q_in, 0, 0, HH, ROWS_Q, HH, HH, 1, 1, 1.f, 1.f, 0, 0, 0);
    // FFN
    LN(ybuf, bp.ln_f_g, bp.ln_f_b, hn, ROWS_Q);
    CVT(P(bp.W1), bfW, (long long)FFD * HH);
    gemm(hn, 0, 0, HH, bfW, 0, 0, HH, ff1, 0, 0, FFD, P(bp.b1),
         nullptr, 0, 0, 0, ROWS_Q, FFD, HH, 1, 1, 1.f, 0.f, 1, 1, 0);
    CVT(P(bp.W2), bfW, (long long)HH * FFD);
    gemm(ff1, 0, 0, FFD, bfW, 0, 0, FFD, out, 0, 0, HH, P(bp.b2),
         ybuf, 0, 0, HH, ROWS_Q, HH, FFD, 1, 1, 1.f, 1.f, 0, 0, 0);
  };

  for (int s = 0; s < 3; ++s) {
    const StageIdx& st = sp[s];
    float* q_next = (s & 1) ? qB : qA;

    // ---- ew: cross-attn over sentence -----------------------------------
    run_std_block(st.ew, cur_q, P(iSent), false, NNW, I(iSentMask), 0, ew_out);
    // ---- se: masked self-attn -------------------------------------------
    run_std_block(st.se, cur_q, nullptr, true, MKD, I(iEntMask), 1, se_out);
    zero_rows_k<<<dim3((unsigned)((QN + 255) / 256)), 256, 0, stream>>>(
        se_out, single, MKD, HH, ROWS_Q);

    // ---- me: mention attention over K=32 candidates ---------------------
    {
      const BPidx& bp = st.me;
      LN(mention, bp.ln_q_g, bp.ln_q_b, me_qn, BV * MMW);
      LN(cur_q, bp.ln_x_g, bp.ln_x_b, xn, ROWS_Q);   // key_mention = query reshaped
      CVT(P(bp.Wi), bfW, (long long)3 * HH * HH);
      const float* bi = P(bp.bi);
      gemm(me_qn, 0, 0, HH, bfW, 0, 0, HH, me_Q, 0, 0, HH, bi,
           nullptr, 0, 0, 0, BV * MMW, HH, HH, 1, 1, 1.f, 0.f, 0, 1, 0);
      gemm(xn, 0, 0, HH, bfW + (size_t)HH * HH, 0, 0, HH, Kp, 0, 0, HH, bi + HH,
           nullptr, 0, 0, 0, ROWS_Q, HH, HH, 1, 1, 1.f, 0.f, 0, 1, 0);
      gemm(xn, 0, 0, HH, bfW + (size_t)2 * HH * HH, 0, 0, HH, Vp, 0, 0, HH, bi + 2 * HH,
           nullptr, 0, 0, 0, ROWS_Q, HH, HH, 1, 1, 1.f, 0.f, 0, 1, 0);
      me_attn_k<<<dim3(BV * MMW * NHEADS), 32, 0, stream>>>(me_Q, Kp, Vp, padAdj, me_O);
      CVT(P(bp.Wo), bfW, (long long)HH * HH);
      gemm(me_O, 0, 0, HH, bfW, 0, 0, HH, me_y, 0, 0, HH, P(bp.bo),
           mention, 0, 0, HH, BV * MMW, HH, HH, 1, 1, 1.f, 1.f, 0, 0, 0);
      LN(me_y, bp.ln_f_g, bp.ln_f_b, me_hn, BV * MMW);
      CVT(P(bp.W1), bfW, (long long)FFD * HH);
      gemm(me_hn, 0, 0, HH, bfW, 0, 0, HH, me_ff1, 0, 0, FFD, P(bp.b1),
           nullptr, 0, 0, 0, BV * MMW, FFD, HH, 1, 1, 1.f, 0.f, 1, 1, 0);
      CVT(P(bp.W2), bfW, (long long)HH * FFD);
      gemm(me_ff1, 0, 0, FFD, bfW, 0, 0, FFD, me_out, 0, 0, HH, P(bp.b2),
           me_y, 0, 0, HH, BV * MMW, HH, FFD, 1, 1, 1.f, 1.f, 0, 0, 0);
      zero_rows_k<<<dim3((unsigned)((MEN + 255) / 256)), 256, 0, stream>>>(
          me_out, allPad, 1, HH, BV * MMW);
    }

    // ---- combine: ctx = ln(ew)+ln(se)+ln(me broadcast) ------------------
    norm3_sum_k<<<dim3((unsigned)ROWS_Q), 256, 0, stream>>>(
        ew_out, se_out, me_out, KKW, P(st.comb_g), P(st.comb_b), ctx, HH);

    // ---- ctx_kg = 0.5*(ctx + kg @ ctx), batched over b ------------------
    cvtT_bf16_k<<<dim3((unsigned)((QN + 255) / 256)), 256, 0, stream>>>(
        ctx, ctxT, MKD, HH, (long long)QN);          // ctxT[b][hcol][row]
    gemm(kg_bf, (long long)MKD * MKD, 0, MKD,
         ctxT, (long long)HH * MKD, 0, MKD,
         q_next, (long long)MKD * HH, 0, HH,
         nullptr, ctx, (long long)MKD * HH, 0, HH,
         MKD, HH, MKD, BV, 1, 0.5f, 0.5f, 0, 0, 0);

    // ---- prediction heads (s < 2) ---------------------------------------
    if (s < 2) {
      CVT(ctx, ctx_bf, (long long)QN);
      CVT(P(st.p_W0), bfW, (long long)HH * HH);
      gemm(ctx_bf, 0, 0, HH, bfW, 0, 0, HH, t0, 0, 0, HH, P(st.p_b0),
           nullptr, 0, 0, 0, ROWS_Q, HH, HH, 1, 1, 1.f, 0.f, 1, 0, 0);
      rowdot_k<<<dim3(ROWS_Q / 8), 256, 0, stream>>>(t0, P(st.p_W1), P(st.p_b1),
                                                     sc0, ROWS_Q, HH);
      CVT(q_next, ckg_bf, (long long)QN);
      gemm(ckg_bf, 0, 0, HH, bfW, 0, 0, HH, t0, 0, 0, HH, P(st.p_b0),
           nullptr, 0, 0, 0, ROWS_Q, HH, HH, 1, 1, 1.f, 0.f, 1, 0, 0);
      rowdot_k<<<dim3(ROWS_Q / 8), 256, 0, stream>>>(t0, P(st.p_W1), P(st.p_b1),
                                                     sc1, ROWS_Q, HH);
      float* outScores = (float*)d_out + 2 * QN + (size_t)s * ROWS_Q;
      max2_k<<<dim3((ROWS_Q + 255) / 256), 256, 0, stream>>>(sc0, sc1, outScores, ROWS_Q);
    }
    cur_q = q_next;   // query = ctx_kg
  }

  // ---- final outputs: ctx, ctx_kg ----
  hipMemcpyAsync((float*)d_out, ctx, QN * sizeof(float),
                 hipMemcpyDeviceToDevice, stream);
  hipMemcpyAsync((float*)d_out + QN, cur_q, QN * sizeof(float),
                 hipMemcpyDeviceToDevice, stream);
}
